// LinearAttention_83141976917054
// MI455X (gfx1250) — matrix operand order repaired and checked by reading, not compiled, not run on hardware
//
#include <hip/hip_runtime.h>
#include <stdint.h>

typedef __bf16 bf16_t;
typedef __attribute__((ext_vector_type(16))) __bf16 v16bf;
typedef __attribute__((ext_vector_type(8)))  float  v8f;

// Problem constants fixed by the reference
constexpr int Bn = 4, S = 4096, D = 1024, H = 16, Hd = 64;
constexpr int NS = Bn * S;      // 16384 rows
constexpr int HD = H * Hd;      // 1024
constexpr float EPSV = 1e-6f;

union FragU { v16bf v; uint32_t u[8]; bf16_t h[16]; };
union AccU  { v8f v; float f[8]; };
union U32B  { uint32_t u; bf16_t h[2]; };

__device__ __forceinline__ float eluplus1(float x) {
  return x > 0.f ? x + 1.f : __expf(x);
}

// A-frag K order used here (contiguous, same as B): lanes 0-15 hold K {0..15},
// lanes 16-31 hold K {16..31}; reg i covers a packed K pair.
__device__ __forceinline__ int a_k_of(int i, int hi) {
  return 2 * i + (hi ? 16 : 0);
}

// Fragment gather for a row-major (m,k) tile: element (m,k) at base[m*ldE + k].
__device__ __forceinline__ v16bf load_frag_rowmajor(const bf16_t* base, int ldE, int lane) {
  FragU r;
  const int hi = lane >> 4;
  const bf16_t* rowp = base + (size_t)(lane & 15) * ldE;
#pragma unroll
  for (int i = 0; i < 8; ++i) {
    r.u[i] = *(const uint32_t*)(rowp + a_k_of(i, hi));
  }
  return r.v;
}

// B-frag K order used here (contiguous, same as A): reg j, lanes 0-15 -> K=2j,2j+1 ; lanes 16-31 ->
// K=16+2j,17+2j ; lane&15 = n. Tile is staged column-major: base[n*ldE + k].
__device__ __forceinline__ v16bf load_fragB_colmajor(const bf16_t* base, int ldE, int lane) {
  FragU r;
  const int hi = lane >> 4;
  const bf16_t* colp = base + (size_t)(lane & 15) * ldE + (hi ? 16 : 0);
#pragma unroll
  for (int j = 0; j < 8; ++j) {
    r.u[j] = *(const uint32_t*)(colp + 2 * j);
  }
  return r.v;
}

__device__ __forceinline__ v8f wmma_bf16(v16bf a, v16bf b, v8f c) {
  return __builtin_amdgcn_wmma_f32_16x16x32_bf16(false, a, false, b, (short)0, c,
                                                 false, false);
}

// ---------------------------------------------------------------------------
// fp32 -> bf16 conversion with optional column restriding (packs Wq|Wk|Wv).
// ---------------------------------------------------------------------------
__global__ void cvt_kernel(const float* __restrict__ src, bf16_t* __restrict__ dst,
                           int rows, int cols, int dstLd) {
  int idx = blockIdx.x * blockDim.x + threadIdx.x;
  if (idx >= rows * cols) return;
  int r = idx / cols, c = idx - r * cols;
  dst[(size_t)r * dstLd + c] = (bf16_t)src[idx];
}

// ---------------------------------------------------------------------------
// Shared WMMA GEMM: C[M,N] = A[M,K]*B[K,N], block tile 128x128, K-step 32.
// 8 waves in a 4x2 grid; each wave owns a 32x64 strip = 8 WMMA tiles.
// MODE 0: N=3072 fused QKV -> bias + elu+1 -> bf16 qf/kf/v buffers
// MODE 1: N=1024 output projection -> +bias -> fp32 out
// ---------------------------------------------------------------------------
template <int MODE>
__global__ __launch_bounds__(256) void gemm_bf16_kernel(
    const bf16_t* __restrict__ A, const bf16_t* __restrict__ Bm, int K, int N,
    bf16_t* __restrict__ qf, bf16_t* __restrict__ kf, bf16_t* __restrict__ vb,
    const float* __restrict__ bq, const float* __restrict__ bk,
    const float* __restrict__ bv, float* __restrict__ out,
    const float* __restrict__ bo) {
  constexpr int TM = 128, TN = 128, TK = 32, LD = TK + 2;
  __shared__ bf16_t ldsA[TM * LD];  // row-major [m][k]
  __shared__ bf16_t ldsB[TN * LD];  // col-major [n][k]

  const int tid = threadIdx.x;
  const int lane = tid & 31, wave = tid >> 5;
  const int waveM = wave >> 1;   // 0..3 -> 32-row strip
  const int waveN = wave & 1;    // 0..1 -> 64-col strip
  const int rowBase = blockIdx.x * TM;
  const int colBase = blockIdx.y * TN;

  v8f acc[2][4] = {};

  for (int kb = 0; kb < K; kb += TK) {
    // Stage A (128x32): u32 loads of packed K pairs, row-major in LDS.
    for (int i = tid; i < TM * (TK / 2); i += 256) {
      int r = i >> 4, kp = i & 15;
      uint32_t w = *(const uint32_t*)(A + (size_t)(rowBase + r) * K + kb + 2 * kp);
      *(uint32_t*)(&ldsA[r * LD + 2 * kp]) = w;
    }
    // Stage B (32x128) transposed -> col-major [n][k] so frag reads are b32.
    for (int i = tid; i < TK * (TN / 2); i += 256) {
      int kk = i >> 6, np = i & 63;
      U32B t;
      t.u = *(const uint32_t*)(Bm + (size_t)(kb + kk) * N + colBase + 2 * np);
      ldsB[(2 * np) * LD + kk]     = t.h[0];
      ldsB[(2 * np + 1) * LD + kk] = t.h[1];
    }
    // Prefetch next K-tile into cache while WMMAs run (global_prefetch_b8).
    if (kb + TK < K) {
      __builtin_prefetch(A + (size_t)(rowBase + (tid >> 4)) * K + (kb + TK) +
                             2 * (tid & 15), 0, 3);
      __builtin_prefetch(Bm + (size_t)(kb + TK + (tid >> 6)) * N + colBase +
                             2 * (tid & 63), 0, 3);
    }
    __syncthreads();

    v16bf af0 = load_frag_rowmajor(&ldsA[(waveM * 32) * LD], LD, lane);
    v16bf af1 = load_frag_rowmajor(&ldsA[(waveM * 32 + 16) * LD], LD, lane);
#pragma unroll
    for (int t = 0; t < 4; ++t) {
      v16bf bfrag =
          load_fragB_colmajor(&ldsB[(waveN * 64 + t * 16) * LD], LD, lane);
      acc[0][t] = wmma_bf16(af0, bfrag, acc[0][t]);
      acc[1][t] = wmma_bf16(af1, bfrag, acc[1][t]);
    }
    __syncthreads();
  }

  // Epilogue. C layout: reg r -> M = r + (lane>=16 ? 8 : 0), N = lane&15.
  const int hi = lane >> 4;
#pragma unroll
  for (int mt = 0; mt < 2; ++mt) {
#pragma unroll
    for (int t = 0; t < 4; ++t) {
      AccU a; a.v = acc[mt][t];
#pragma unroll
      for (int r = 0; r < 8; ++r) {
        int row = rowBase + waveM * 32 + mt * 16 + (hi ? 8 : 0) + r;
        int col = colBase + waveN * 64 + t * 16 + (lane & 15);
        float v = a.f[r];
        if (MODE == 0) {
          if (col < D) {
            qf[(size_t)row * D + col] = (bf16_t)eluplus1(v + bq[col]);
          } else if (col < 2 * D) {
            kf[(size_t)row * D + (col - D)] = (bf16_t)eluplus1(v + bk[col - D]);
          } else {
            vb[(size_t)row * D + (col - 2 * D)] = (bf16_t)(v + bv[col - 2 * D]);
          }
        } else {
          out[(size_t)row * N + col] = v + bo[col];
        }
      }
    }
  }
}

// ---------------------------------------------------------------------------
// Per-(n,h): KV[m,d] = sum_s v[s,m]*kf[s,d] (M=N=64, K=4096) and
// ksum[d] = sum_s kf[s,d]. Stores KV transposed (d-major) as bf16.
// Waves: (mtile = w&3, khalf = w>>2); khalf partials reduced through LDS.
// ---------------------------------------------------------------------------
__global__ __launch_bounds__(256) void kv_kernel(const bf16_t* __restrict__ kf,
                                                 const bf16_t* __restrict__ vb,
                                                 bf16_t* __restrict__ kvT,
                                                 float* __restrict__ ksum) {
  constexpr int LD = 34;
  __shared__ bf16_t ldsKT[2][64 * LD];  // kf tile, col-major [d][k]
  __shared__ bf16_t ldsVT[2][64 * LD];  // v tile,  col-major [m][k]
  __shared__ float ldsRed[64 * 64];

  const int nb = blockIdx.x;  // n*16 + h
  const int n = nb >> 4, h = nb & 15;
  const int tid = threadIdx.x, lane = tid & 31, wave = tid >> 5;
  const int mtile = wave & 3, khalf = wave >> 2;
  const size_t rowOff = (size_t)n * S;
  const int colOff = h * Hd;

  // ---- ksum (fp32) ----
  {
    int d = tid & 63, part = tid >> 6;  // 4 partial sums per d
    float a = 0.f;
    for (int s = part * (S / 4); s < (part + 1) * (S / 4); ++s)
      a += (float)kf[(rowOff + s) * HD + colOff + d];
    ldsRed[part * 64 + d] = a;
  }
  __syncthreads();
  if (tid < 64) {
    float a = ldsRed[tid] + ldsRed[64 + tid] + ldsRed[128 + tid] + ldsRed[192 + tid];
    ksum[(size_t)nb * Hd + tid] = a;
  }
  __syncthreads();

  // ---- KV accumulation ----
  v8f acc[4] = {};
  for (int kb = 0; kb < S / 2; kb += 32) {
    for (int i = tid; i < 2 * 32 * 32; i += 256) {
      int half = i >> 10, rem = i & 1023;
      int kk = rem >> 5, np = rem & 31;
      size_t grow = rowOff + (size_t)half * (S / 2) + kb + kk;
      U32B tk, tv;
      tk.u = *(const uint32_t*)(kf + grow * HD + colOff + 2 * np);
      tv.u = *(const uint32_t*)(vb + grow * HD + colOff + 2 * np);
      ldsKT[half][(2 * np) * LD + kk]     = tk.h[0];
      ldsKT[half][(2 * np + 1) * LD + kk] = tk.h[1];
      ldsVT[half][(2 * np) * LD + kk]     = tv.h[0];
      ldsVT[half][(2 * np + 1) * LD + kk] = tv.h[1];
    }
    __syncthreads();
    // A[m,k] = v[s=kb+k, m] -> row-major view of col-major ldsVT
    v16bf afrag = load_frag_rowmajor(&ldsVT[khalf][(mtile * 16) * LD], LD, lane);
#pragma unroll
    for (int t = 0; t < 4; ++t) {
      v16bf bfrag = load_fragB_colmajor(&ldsKT[khalf][(t * 16) * LD], LD, lane);
      acc[t] = wmma_bf16(afrag, bfrag, acc[t]);
    }
    __syncthreads();
  }

  // ---- reduce the two K-halves, store kvT[d][m] bf16 ----
  const int hi = lane >> 4;
  if (khalf == 0) {
#pragma unroll
    for (int t = 0; t < 4; ++t) {
      AccU a; a.v = acc[t];
#pragma unroll
      for (int r = 0; r < 8; ++r) {
        int m = mtile * 16 + (hi ? 8 : 0) + r;
        int d = t * 16 + (lane & 15);
        ldsRed[m * 64 + d] = a.f[r];
      }
    }
  }
  __syncthreads();
  if (khalf == 1) {
#pragma unroll
    for (int t = 0; t < 4; ++t) {
      AccU a; a.v = acc[t];
#pragma unroll
      for (int r = 0; r < 8; ++r) {
        int m = mtile * 16 + (hi ? 8 : 0) + r;
        int d = t * 16 + (lane & 15);
        float v = a.f[r] + ldsRed[m * 64 + d];
        kvT[(size_t)nb * (Hd * Hd) + d * 64 + m] = (bf16_t)v;
      }
    }
  }
}

// ---------------------------------------------------------------------------
// z[n,l,h] = 1 / (sum_d qf[n,l,h,d] * ksum[n,h,d] + eps)
// ---------------------------------------------------------------------------
__global__ void z_kernel(const bf16_t* __restrict__ qf,
                         const float* __restrict__ ksum,
                         float* __restrict__ zbuf) {
  int idx = blockIdx.x * blockDim.x + threadIdx.x;  // (n*S + l)*H + h
  if (idx >= NS * H) return;
  int h = idx & 15;
  int row = idx >> 4;       // n*S + l
  int n = row >> 12;        // / S
  const bf16_t* qp = qf + (size_t)row * HD + h * Hd;
  const float* kp = ksum + (size_t)(n * H + h) * Hd;
  float a = 0.f;
#pragma unroll
  for (int d = 0; d < Hd; ++d) a += (float)qp[d] * kp[d];
  zbuf[idx] = 1.f / (a + EPSV);
}

// ---------------------------------------------------------------------------
// y[l,m] = z[l] * sum_d qf[l,d] * KV[m,d]  per (n,h); M=l tile, N=m, K=d=64.
// One block per (n,h,l-chunk of 128); KV tile cached in LDS.
// ---------------------------------------------------------------------------
__global__ __launch_bounds__(256) void y_kernel(const bf16_t* __restrict__ qf,
                                                const bf16_t* __restrict__ kvT,
                                                const float* __restrict__ zbuf,
                                                bf16_t* __restrict__ ybuf) {
  constexpr int LD = 66;
  __shared__ bf16_t ldsKV[64 * LD];  // col-major [m][d]

  int bid = blockIdx.x;
  int lchunk = bid & 31;  // S/128 = 32
  int nh = bid >> 5;
  int n = nh >> 4, h = nh & 15;
  int tid = threadIdx.x, lane = tid & 31, wave = tid >> 5;

  // Stage kvT (row d, col m) -> LDS col-major [m][d]
  for (int i = tid; i < 64 * 32; i += 256) {
    int dd = i >> 5, mp = i & 31;
    U32B t;
    t.u = *(const uint32_t*)(kvT + (size_t)nh * (Hd * Hd) + dd * 64 + 2 * mp);
    ldsKV[(2 * mp) * LD + dd]     = t.h[0];
    ldsKV[(2 * mp + 1) * LD + dd] = t.h[1];
  }
  __syncthreads();

  const int lbase = lchunk * 128 + wave * 16;
  const size_t arow = (size_t)n * S + lbase;

  v8f acc[4] = {};
#pragma unroll
  for (int ks = 0; ks < 2; ++ks) {
    // A frag straight from global: qf is row-major with d contiguous.
    v16bf afrag =
        load_frag_rowmajor(qf + arow * HD + h * Hd + ks * 32, HD, lane);
#pragma unroll
    for (int t = 0; t < 4; ++t) {
      v16bf bfrag = load_fragB_colmajor(&ldsKV[(t * 16) * LD + ks * 32], LD, lane);
      acc[t] = wmma_bf16(afrag, bfrag, acc[t]);
    }
  }

  const int hi = lane >> 4;
  float zr[8];
#pragma unroll
  for (int r = 0; r < 8; ++r) {
    int l = lbase + (hi ? 8 : 0) + r;
    zr[r] = zbuf[((size_t)n * S + l) * H + h];
  }
#pragma unroll
  for (int t = 0; t < 4; ++t) {
    AccU a; a.v = acc[t];
#pragma unroll
    for (int r = 0; r < 8; ++r) {
      int l = lbase + (hi ? 8 : 0) + r;
      int m = t * 16 + (lane & 15);
      ybuf[((size_t)n * S + l) * HD + h * Hd + m] = (bf16_t)(a.f[r] * zr[r]);
    }
  }
}

// ---------------------------------------------------------------------------
extern "C" void kernel_launch(void* const* d_in, const int* in_sizes, int n_in,
                              void* d_out, int out_size, void* d_ws,
                              size_t ws_size, hipStream_t stream) {
  (void)in_sizes; (void)n_in; (void)out_size; (void)ws_size;
  const float* x  = (const float*)d_in[0];
  const float* Wq = (const float*)d_in[1];
  const float* bq = (const float*)d_in[2];
  const float* Wk = (const float*)d_in[3];
  const float* bk = (const float*)d_in[4];
  const float* Wv = (const float*)d_in[5];
  const float* bv = (const float*)d_in[6];
  const float* Wo = (const float*)d_in[7];
  const float* bo = (const float*)d_in[8];
  float* out = (float*)d_out;

  char* ws = (char*)d_ws;
  size_t off = 0;
  auto alloc = [&](size_t bytes) {
    void* p = ws + off;
    off = (off + bytes + 255) & ~(size_t)255;
    return p;
  };
  bf16_t* xb   = (bf16_t*)alloc((size_t)NS * D * 2);
  bf16_t* wqkv = (bf16_t*)alloc((size_t)D * 3 * HD * 2);
  bf16_t* wob  = (bf16_t*)alloc((size_t)HD * D * 2);
  bf16_t* qf   = (bf16_t*)alloc((size_t)NS * HD * 2);
  bf16_t* kfb  = (bf16_t*)alloc((size_t)NS * HD * 2);
  bf16_t* vbb  = (bf16_t*)alloc((size_t)NS * HD * 2);
  bf16_t* kvt  = (bf16_t*)alloc((size_t)Bn * H * Hd * Hd * 2);
  float*  ksum = (float*)alloc((size_t)Bn * H * Hd * 4);
  float*  zbuf = (float*)alloc((size_t)NS * H * 4);
  bf16_t* ybuf = xb;  // xb dead after the QKV GEMM; reuse for y

  // bf16 conversions
  cvt_kernel<<<(NS * D + 255) / 256, 256, 0, stream>>>(x, xb, NS, D, D);
  cvt_kernel<<<(D * HD + 255) / 256, 256, 0, stream>>>(Wq, wqkv, D, HD, 3 * HD);
  cvt_kernel<<<(D * HD + 255) / 256, 256, 0, stream>>>(Wk, wqkv + HD, D, HD, 3 * HD);
  cvt_kernel<<<(D * HD + 255) / 256, 256, 0, stream>>>(Wv, wqkv + 2 * HD, D, HD, 3 * HD);
  cvt_kernel<<<(HD * D + 255) / 256, 256, 0, stream>>>(Wo, wob, HD, D, D);

  // Fused QKV projection + feature map
  dim3 g0(NS / 128, (3 * HD) / 128);
  gemm_bf16_kernel<0><<<g0, 256, 0, stream>>>(xb, wqkv, D, 3 * HD, qf, kfb, vbb,
                                              bq, bk, bv, nullptr, nullptr);

  // KV state + ksum
  kv_kernel<<<Bn * H, 256, 0, stream>>>(kfb, vbb, kvt, ksum);

  // Normalizer
  z_kernel<<<(NS * H + 255) / 256, 256, 0, stream>>>(qf, ksum, zbuf);

  // Attention output y
  y_kernel<<<Bn * H * (S / 128), 256, 0, stream>>>(qf, kvt, zbuf, ybuf);

  // Output projection -> fp32 out
  dim3 g1(NS / 128, D / 128);
  gemm_bf16_kernel<1><<<g1, 256, 0, stream>>>(ybuf, wob, HD, D, nullptr, nullptr,
                                              nullptr, nullptr, nullptr, nullptr,
                                              out, bo);
}
